// PFNet_59287728554030
// MI455X (gfx1250) — matrix operand order repaired and checked by reading, not compiled, not run on hardware
//
#include <hip/hip_runtime.h>
#include <hip/hip_bf16.h>

typedef __attribute__((ext_vector_type(16))) _Float16 v16h;
typedef __attribute__((ext_vector_type(8)))  _Float16 v8h;
typedef __attribute__((ext_vector_type(8)))  float    v8f;

#define BB    4
#define NN    5000
#define MM    (BB*NN)
#define NB_   10
#define CHK   500
#define DP_   282
#define HH    256

__device__ __forceinline__ float seluf(float x){
  const float a = 1.6732632423543772f, s = 1.0507009873554805f;
  return x > 0.f ? s * x : s * (a * (expf(x) - 1.f));
}

// ---------------- weight prep: fp32 [K,N] -> fp16 transposed [Npad,Kpad] ----------
__global__ void k_wprep(const float* __restrict__ W, _Float16* __restrict__ Wt,
                        int K, int Nn, int Kpad, int Npad){
  int i = blockIdx.x*blockDim.x + threadIdx.x;
  if (i >= Npad*Kpad) return;
  int n = i / Kpad, k = i - n*Kpad;
  float v = (n < Nn && k < K) ? W[(size_t)k*Nn + n] : 0.f;
  Wt[i] = (_Float16)v;
}

__global__ void k_bprep(const float* __restrict__ b, float* __restrict__ bp, int Nn, int Npad){
  int i = blockIdx.x*blockDim.x + threadIdx.x;
  if (i < Npad) bp[i] = (i < Nn) ? b[i] : 0.f;
}

__global__ void k_zero_i(int* p, int n){
  int i = blockIdx.x*blockDim.x + threadIdx.x; if (i < n) p[i] = 0;
}

// ---------------- encode: one-hot(12) + features; fill concat pads ----------------
__global__ void k_encode(const float* __restrict__ X, float* __restrict__ pts,
                         _Float16* __restrict__ enc16, float* __restrict__ msk,
                         _Float16* __restrict__ dec16, _Float16* __restrict__ xr16){
  int i = blockIdx.x*blockDim.x + threadIdx.x;
  if (i >= MM) return;
  const float* x = X + (size_t)i*15;
  int cls = (int)x[0];
  float e[26];
  for (int j = 0; j < 12; ++j) e[j] = (j == cls) ? 1.f : 0.f;
  for (int j = 0; j < 14; ++j) e[12+j] = x[1+j];
  float* pr = pts + (size_t)i*DP_;
  for (int j = 0; j < 26; ++j) pr[j] = e[j];
  _Float16* er = enc16 + (size_t)i*32;
  for (int j = 0; j < 32; ++j) er[j] = (j < 26) ? (_Float16)e[j] : (_Float16)0.f;
  _Float16* dr = dec16 + (size_t)i*288;
  for (int j = 0; j < 26; ++j) dr[j] = (_Float16)e[j];
  for (int j = 282; j < 288; ++j) dr[j] = (_Float16)0.f;
  _Float16* xr = xr16 + (size_t)i*320;
  for (int j = 0; j < 26; ++j) xr[j] = (_Float16)e[j];
  for (int j = 290; j < 320; ++j) xr[j] = (_Float16)0.f;
  msk[i] = (x[0] != 0.f) ? 1.f : 0.f;
}

// ---------------- WMMA GEMM: out = act(A16[M,Kpad] @ Bt16[Nn,Kpad]^T + bias) ------
// Each wave computes a 16 x (16*NT) strip: one A fragment reused across NT B tiles.
// Loads for all NT B fragments are issued first (single clause), then NT WMMAs
// back-to-back, so wmma[n] overlaps the outstanding loads of fragments n+1..NT-1.
// A-fragment: lane m=lane&15, K=half*16+e  -> contiguous in row-major A
// B-fragment: lane n=lane&15, K=half*16+e  -> contiguous in transposed weights
// C/D: VGPR r -> row = r + 8*half, col = lane&15
template<int NT>
__global__ void k_gemm_t(const _Float16* __restrict__ A, int lda,
                         const _Float16* __restrict__ Bt, int ldb,
                         const float* __restrict__ bias,
                         float* __restrict__ out, int ldo, int ocol,
                         _Float16* __restrict__ out16, int ldo16, int ocol16,
                         int M, int Nn, int Kpad, int act){
  int wid = (blockIdx.x*blockDim.x + threadIdx.x) >> 5;
  int groups_n = (Nn >> 4) / NT;
  int tm = wid / groups_n, tg = wid - tm*groups_n;
  if (tm*16 >= M) return;                      // uniform per wave -> EXEC all-1s
  int lane = threadIdx.x & 31;
  int hlf = lane >> 4, q = lane & 15;
  const _Float16* arow = A + (size_t)(tm*16 + q) * lda;
  const _Float16* brow[NT];
#pragma unroll
  for (int nt = 0; nt < NT; ++nt)
    brow[nt] = Bt + (size_t)(tg*NT*16 + nt*16 + q) * ldb;
  v8f acc[NT];
#pragma unroll
  for (int nt = 0; nt < NT; ++nt) acc[nt] = (v8f){0.f,0.f,0.f,0.f,0.f,0.f,0.f,0.f};
  for (int kk = 0; kk < Kpad; kk += 32){
    v16h a = *(const v16h*)(arow + kk + hlf*16);
    v16h bfrag[NT];
#pragma unroll
    for (int nt = 0; nt < NT; ++nt)
      bfrag[nt] = *(const v16h*)(brow[nt] + kk + hlf*16);
    if (kk + 32 < Kpad) __builtin_prefetch(arow + kk + 64, 0, 1);
#pragma unroll
    for (int nt = 0; nt < NT; ++nt)
      acc[nt] = __builtin_amdgcn_wmma_f32_16x16x32_f16(false, a, false, bfrag[nt],
                                                       (short)0, acc[nt], false, false);
  }
#pragma unroll
  for (int nt = 0; nt < NT; ++nt){
    int col = tg*NT*16 + nt*16 + q;
    float bv = bias ? bias[col] : 0.f;
#pragma unroll
    for (int r = 0; r < 8; ++r){
      int row = tm*16 + hlf*8 + r;
      float v = acc[nt][r] + bv;
      if (act == 1) v = seluf(v);
      else if (act == 2) v = 1.f / (1.f + expf(-v));
      if (out)   out  [(size_t)row*ldo  + ocol  + col] = v;
      if (out16) out16[(size_t)row*ldo16 + ocol16 + col] = (_Float16)v;
    }
  }
}

// ---------------- LSH binning ----------------
__global__ void k_binmul(const float* __restrict__ pts, const float* __restrict__ rot,
                         int* __restrict__ bin_idx){
  int i = blockIdx.x*blockDim.x + threadIdx.x;
  if (i >= MM) return;
  const float* p = pts + (size_t)i*DP_;
  float m0=0,m1=0,m2=0,m3=0,m4=0;
  for (int k = 0; k < DP_; ++k){
    float v = p[k];
    const float* r = rot + (size_t)k*5;
    m0 += v*r[0]; m1 += v*r[1]; m2 += v*r[2]; m3 += v*r[3]; m4 += v*r[4];
  }
  float mv[10] = {m0,m1,m2,m3,m4,-m0,-m1,-m2,-m3,-m4};
  int bi = 0; float bv = mv[0];
  for (int r = 1; r < 10; ++r) if (mv[r] > bv){ bv = mv[r]; bi = r; }
  bin_idx[i] = bi;
}

__global__ void k_hist(const int* __restrict__ bin_idx, int* __restrict__ hist){
  int i = blockIdx.x*blockDim.x + threadIdx.x;
  if (i >= MM) return;
  int b = i / NN;
  atomicAdd(&hist[b*NB_ + bin_idx[i]], 1);
}

__global__ void k_offsets(const int* __restrict__ hist, int* __restrict__ offs){
  int b = threadIdx.x;
  if (b >= BB) return;
  int acc = 0;
  for (int g = 0; g < NB_; ++g){ offs[b*NB_+g] = acc; acc += hist[b*NB_+g]; }
}

// stable counting-sort compaction == stable argsort by bin
__global__ void k_compact(const int* __restrict__ bin_idx, const int* __restrict__ offs,
                          int* __restrict__ perm){
  int b = blockIdx.x / NB_, g = blockIdx.x % NB_;
  const int* bi = bin_idx + (size_t)b*NN;
  int base = offs[b*NB_+g];
  __shared__ int s[256];
  int t = threadIdx.x;
  for (int c = 0; c < NN; c += 256){
    int i = c + t;
    int flag = (i < NN && bi[i] == g) ? 1 : 0;
    s[t] = flag; __syncthreads();
    for (int o = 1; o < 256; o <<= 1){
      int v = (t >= o) ? s[t-o] : 0;
      __syncthreads();
      s[t] += v;
      __syncthreads();
    }
    if (flag) perm[(size_t)b*NN + base + s[t] - 1] = i;
    base += s[255];
    __syncthreads();
  }
}

// ---------------- gather + l2norm to fp16 padded tiles [gb][512][288] -------------
__global__ void k_gather_norm(const float* __restrict__ pts, const int* __restrict__ perm,
                              _Float16* __restrict__ nrm){
  int wid = (blockIdx.x*blockDim.x + threadIdx.x) >> 5;
  int lane = threadIdx.x & 31;
  int gb = wid >> 9;
  int r  = wid & 511;
  _Float16* dst = nrm + ((size_t)gb*512 + r)*288;
  if (r >= CHK){
    for (int k = lane; k < 288; k += 32) dst[k] = (_Float16)0.f;
    return;
  }
  int b = gb / NB_, g = gb - b*NB_;
  int src = perm[(size_t)b*NN + g*CHK + r];
  const float* p = pts + ((size_t)b*NN + src)*DP_;
  float ss = 0.f;
  for (int k = lane; k < DP_; k += 32){ float v = p[k]; ss += v*v; }
  for (int o = 16; o; o >>= 1) ss += __shfl_xor(ss, o, 32);
  float rn = rsqrtf(fmaxf(ss, 1e-12f));
  for (int k = lane; k < 288; k += 32)
    dst[k] = (k < DP_) ? (_Float16)(p[k]*rn) : (_Float16)0.f;
}

// ---------------- wave-per-row softmax + top-5 (tie -> lowest index) --------------
__global__ void k_topk(const float* __restrict__ dm, const int* __restrict__ perm,
                       int* __restrict__ cols, float* __restrict__ vals){
  int wid = (blockIdx.x*blockDim.x + threadIdx.x) >> 5;
  int lane = threadIdx.x & 31;
  int gb = wid / CHK;
  int r  = wid - gb*CHK;
  if (gb >= BB*NB_) return;
  const float* s = dm + ((size_t)gb*512 + r)*512;
  float mx = -3.4e38f;
  for (int j = lane; j < CHK; j += 32) mx = fmaxf(mx, s[j]);
  for (int o = 16; o; o >>= 1) mx = fmaxf(mx, __shfl_xor(mx, o, 32));
  float se = 0.f;
  for (int j = lane; j < CHK; j += 32) se += expf(s[j] - mx);
  for (int o = 16; o; o >>= 1) se += __shfl_xor(se, o, 32);
  int b = gb / NB_, g = gb - b*NB_;
  size_t pout = (size_t)b*NN + g*CHK + r;
  int chosen[5];
  for (int kk = 0; kk < 5; ++kk){
    float bv = -3.4e38f; int bi = 0x7fffffff;
    for (int j = lane; j < CHK; j += 32){
      bool skip = false;
      for (int t = 0; t < kk; ++t) skip = skip || (chosen[t] == j);
      if (skip) continue;
      float v = s[j];
      if (v > bv || (v == bv && j < bi)){ bv = v; bi = j; }
    }
    for (int o = 16; o; o >>= 1){
      float ov = __shfl_xor(bv, o, 32);
      int   oi = __shfl_xor(bi, o, 32);
      if (ov > bv || (ov == bv && oi < bi)){ bv = ov; bi = oi; }
    }
    chosen[kk] = bi;
    if (lane == 0){
      vals[pout*5 + kk] = expf(bv - mx) / se;
      cols[pout*5 + kk] = perm[(size_t)b*NN + g*CHK + bi];
    }
  }
}

__global__ void k_degnorm(const int* __restrict__ perm, const float* __restrict__ vals,
                          float* __restrict__ normv){
  int p = blockIdx.x*blockDim.x + threadIdx.x;
  if (p >= MM) return;
  int b = p / NN;
  int row = perm[p];
  float s = 0.f;
  for (int k = 0; k < 5; ++k) s += vals[(size_t)p*5 + k];
  normv[(size_t)b*NN + row] = rsqrtf(s + 1e-6f);
}

// f_hom[perm[p]] = norm[row] * sum_k vals * (t1[cols]*norm[cols])
__global__ void k_spmm(const float* __restrict__ t1, const int* __restrict__ perm,
                       const int* __restrict__ cols, const float* __restrict__ vals,
                       const float* __restrict__ normv, float* __restrict__ fhom){
  int p = blockIdx.x;
  int h = threadIdx.x;
  int b = p / NN;
  int row = perm[p];
  float acc = 0.f;
  for (int k = 0; k < 5; ++k){
    int c = cols[(size_t)p*5 + k];
    acc += vals[(size_t)p*5 + k] * t1[((size_t)b*NN + c)*HH + h] * normv[(size_t)b*NN + c];
  }
  fhom[((size_t)b*NN + row)*HH + h] = acc * normv[(size_t)b*NN + row];
}

__global__ void k_combine(const float* __restrict__ gate, const float* __restrict__ fhom,
                          const float* __restrict__ fhet, _Float16* __restrict__ y16, int n){
  int i = blockIdx.x*blockDim.x + threadIdx.x;
  if (i >= n) return;
  float g = gate[i];
  float v = seluf(g*fhom[i] + (1.f - g)*fhet[i]);
  y16[i] = (_Float16)v;
}

__global__ void k_copy16(const float* __restrict__ src, _Float16* __restrict__ dec16){
  int idx = blockIdx.x*blockDim.x + threadIdx.x;
  if (idx >= MM*HH) return;
  int i = idx >> 8, c = idx & 255;
  dec16[(size_t)i*288 + 26 + c] = (_Float16)src[idx];
}

__global__ void k_fillxr(const float* __restrict__ oid, _Float16* __restrict__ xr16){
  int idx = blockIdx.x*blockDim.x + threadIdx.x;
  if (idx >= MM*8) return;
  int i = idx >> 3, c = idx & 7;
  xr16[(size_t)i*320 + 26 + c] = (_Float16)oid[(size_t)i*16 + c];
}

__global__ void k_final(const float* __restrict__ oid, const float* __restrict__ och,
                        const float* __restrict__ momb, const float* __restrict__ msk,
                        float* __restrict__ out){
  int idx = blockIdx.x*blockDim.x + threadIdx.x;
  if (idx >= MM*12) return;
  int i = idx / 12, c = idx - i*12;
  float v;
  if (c < 8)       v = oid[(size_t)i*16 + c];
  else if (c == 8) v = och[(size_t)i*16];
  else             v = momb[(size_t)i*16 + (c - 9)];
  out[idx] = v * msk[i];
}

// =================================================================================
extern "C" void kernel_launch(void* const* d_in, const int* in_sizes, int n_in,
                              void* d_out, int out_size, void* d_ws, size_t ws_size,
                              hipStream_t stream){
  const float* X          = (const float*)d_in[0];
  const float* W_embed    = (const float*)d_in[1];
  const float* b_embed    = (const float*)d_in[2];
  const float* rot        = (const float*)d_in[3];
  const float* id_enc_W   = (const float*)d_in[4];
  const float* id_enc_b   = (const float*)d_in[5];
  const float* id_Wt      = (const float*)d_in[6];
  const float* id_bt      = (const float*)d_in[7];
  const float* id_Wh      = (const float*)d_in[8];
  const float* id_theta   = (const float*)d_in[9];
  const float* id_dec_W   = (const float*)d_in[10];
  const float* id_dec_b   = (const float*)d_in[11];
  const float* W_id       = (const float*)d_in[12];
  const float* b_id       = (const float*)d_in[13];
  const float* W_charge   = (const float*)d_in[14];
  const float* b_charge   = (const float*)d_in[15];
  const float* reg_enc_W  = (const float*)d_in[16];
  const float* reg_enc_b  = (const float*)d_in[17];
  const float* reg_Wt     = (const float*)d_in[18];
  const float* reg_bt     = (const float*)d_in[19];
  const float* reg_Wh     = (const float*)d_in[20];
  const float* reg_theta  = (const float*)d_in[21];
  const float* reg_dec_W  = (const float*)d_in[22];
  const float* reg_dec_b  = (const float*)d_in[23];
  const float* W_mom      = (const float*)d_in[24];
  const float* b_mom      = (const float*)d_in[25];

  char* w = (char*)d_ws;
  size_t off = 0;
  auto alloc = [&](size_t bytes)->void*{
    void* p = w + off; off = (off + bytes + 255) & ~(size_t)255; return p;
  };

  float*    pts    = (float*)   alloc((size_t)MM*DP_*4);
  _Float16* enc16  = (_Float16*)alloc((size_t)MM*32*2);
  float*    msk    = (float*)   alloc((size_t)MM*4);
  int*      bidx   = (int*)     alloc((size_t)MM*4);
  int*      hist   = (int*)     alloc(40*4);
  int*      offs   = (int*)     alloc(40*4);
  int*      perm   = (int*)     alloc((size_t)MM*4);
  _Float16* nrm    = (_Float16*)alloc((size_t)40*512*288*2);
  float*    dm     = (float*)   alloc((size_t)40*512*512*4);
  int*      cols   = (int*)     alloc((size_t)MM*5*4);
  float*    vals   = (float*)   alloc((size_t)MM*5*4);
  float*    normv  = (float*)   alloc((size_t)MM*4);
  _Float16* x16    = (_Float16*)alloc((size_t)MM*HH*2);
  float*    gate   = (float*)   alloc((size_t)MM*HH*4);
  float*    t1     = (float*)   alloc((size_t)MM*HH*4);
  float*    fhet   = (float*)   alloc((size_t)MM*HH*4);
  float*    fhom   = (float*)   alloc((size_t)MM*HH*4);
  _Float16* y16    = (_Float16*)alloc((size_t)MM*HH*2);
  float*    xid    = (float*)   alloc((size_t)MM*HH*4);
  _Float16* dec16  = (_Float16*)alloc((size_t)MM*288*2);
  _Float16* xr16   = (_Float16*)alloc((size_t)MM*320*2);
  float*    oid    = (float*)   alloc((size_t)MM*16*4);
  float*    och    = (float*)   alloc((size_t)MM*16*4);
  float*    momb   = (float*)   alloc((size_t)MM*16*4);

  _Float16* WembT   = (_Float16*)alloc(256*32*2);
  _Float16* idencT  = (_Float16*)alloc(256*32*2);
  _Float16* idWtT   = (_Float16*)alloc(256*256*2);
  _Float16* idWhT   = (_Float16*)alloc(256*256*2);
  _Float16* idthT   = (_Float16*)alloc(256*256*2);
  _Float16* iddecT  = (_Float16*)alloc(256*256*2);
  _Float16* WidT    = (_Float16*)alloc(16*288*2);
  _Float16* WchT    = (_Float16*)alloc(16*288*2);
  _Float16* regencT = (_Float16*)alloc(256*320*2);
  _Float16* regWtT  = (_Float16*)alloc(256*256*2);
  _Float16* regWhT  = (_Float16*)alloc(256*256*2);
  _Float16* regthT  = (_Float16*)alloc(256*256*2);
  _Float16* regdecT = (_Float16*)alloc(256*256*2);
  _Float16* WmomT   = (_Float16*)alloc(16*288*2);
  float*    bid_p   = (float*)   alloc(16*4);
  float*    bch_p   = (float*)   alloc(16*4);
  float*    bmom_p  = (float*)   alloc(16*4);

  auto wprep = [&](const float* W, _Float16* Wt, int K, int Nn, int Kp, int Np){
    int n = Np*Kp;
    k_wprep<<<(n+255)/256, 256, 0, stream>>>(W, Wt, K, Nn, Kp, Np);
  };
  auto gemm = [&](const _Float16* A, int lda, const _Float16* Bt, int ldb,
                  const float* bias, float* out, int ldo, int ocol,
                  _Float16* o16, int ld16, int oc16, int M_, int Nn_, int Kp_, int act){
    if ((Nn_ & 63) == 0){
      int waves = (M_/16) * ((Nn_/16)/4);
      int blocks = (waves + 7)/8;
      k_gemm_t<4><<<blocks, 256, 0, stream>>>(A, lda, Bt, ldb, bias, out, ldo, ocol,
                                              o16, ld16, oc16, M_, Nn_, Kp_, act);
    } else {
      int waves = (M_/16) * (Nn_/16);
      int blocks = (waves + 7)/8;
      k_gemm_t<1><<<blocks, 256, 0, stream>>>(A, lda, Bt, ldb, bias, out, ldo, ocol,
                                              o16, ld16, oc16, M_, Nn_, Kp_, act);
    }
  };

  // ---- prep ----
  wprep(W_embed,   WembT,   26, 256, 32, 256);
  wprep(id_enc_W,  idencT,  26, 256, 32, 256);
  wprep(id_Wt,     idWtT,  256, 256, 256, 256);
  wprep(id_Wh,     idWhT,  256, 256, 256, 256);
  wprep(id_theta,  idthT,  256, 256, 256, 256);
  wprep(id_dec_W,  iddecT, 256, 256, 256, 256);
  wprep(W_id,      WidT,   282,   8, 288,  16);
  wprep(W_charge,  WchT,   282,   1, 288,  16);
  wprep(reg_enc_W, regencT,290, 256, 320, 256);
  wprep(reg_Wt,    regWtT, 256, 256, 256, 256);
  wprep(reg_Wh,    regWhT, 256, 256, 256, 256);
  wprep(reg_theta, regthT, 256, 256, 256, 256);
  wprep(reg_dec_W, regdecT,256, 256, 256, 256);
  wprep(W_mom,     WmomT,  282,   3, 288,  16);
  k_bprep<<<1, 64, 0, stream>>>(b_id,     bid_p,  8, 16);
  k_bprep<<<1, 64, 0, stream>>>(b_charge, bch_p,  1, 16);
  k_bprep<<<1, 64, 0, stream>>>(b_mom,    bmom_p, 3, 16);
  k_zero_i<<<1, 64, 0, stream>>>(hist, 40);

  // ---- encode + embed (pts = [enc | selu(enc@W_embed+b)]) ----
  k_encode<<<(MM+255)/256, 256, 0, stream>>>(X, pts, enc16, msk, dec16, xr16);
  gemm(enc16, 32, WembT, 32, b_embed, pts, DP_, 26, nullptr, 0, 0, MM, 256, 32, 1);

  // ---- LSH binning + stable sort ----
  k_binmul<<<(MM+255)/256, 256, 0, stream>>>(pts, rot, bidx);
  k_hist<<<(MM+255)/256, 256, 0, stream>>>(bidx, hist);
  k_offsets<<<1, 64, 0, stream>>>(hist, offs);
  k_compact<<<BB*NB_, 256, 0, stream>>>(bidx, offs, perm);

  // ---- gather + normalize, similarity WMMA GEMMs, softmax/top-k ----
  k_gather_norm<<<(BB*NB_*512)/8, 256, 0, stream>>>(pts, perm, nrm);
  for (int gb = 0; gb < BB*NB_; ++gb){
    gemm(nrm + (size_t)gb*512*288, 288, nrm + (size_t)gb*512*288, 288,
         nullptr, dm + (size_t)gb*512*512, 512, 0, nullptr, 0, 0, 512, 512, 288, 0);
  }
  k_topk<<<(BB*NB_*CHK)/8, 256, 0, stream>>>(dm, perm, cols, vals);
  k_degnorm<<<(MM+255)/256, 256, 0, stream>>>(perm, vals, normv);

  // ---- id branch ----
  gemm(enc16, 32, idencT, 32, id_enc_b, nullptr, 0, 0, x16, HH, 0, MM, 256, 32, 1);
  gemm(x16, HH, idWtT, 256, id_bt, gate, HH, 0, nullptr, 0, 0, MM, 256, 256, 2);
  gemm(x16, HH, idthT, 256, nullptr, t1, HH, 0, nullptr, 0, 0, MM, 256, 256, 0);
  gemm(x16, HH, idWhT, 256, nullptr, fhet, HH, 0, nullptr, 0, 0, MM, 256, 256, 0);
  k_spmm<<<MM, HH, 0, stream>>>(t1, perm, cols, vals, normv, fhom);
  k_combine<<<(MM*HH+255)/256, 256, 0, stream>>>(gate, fhom, fhet, y16, MM*HH);
  gemm(y16, HH, iddecT, 256, id_dec_b, xid, HH, 0, xr16, 320, 34, MM, 256, 256, 1);
  k_copy16<<<(MM*HH+255)/256, 256, 0, stream>>>(xid, dec16);
  gemm(dec16, 288, WidT, 288, bid_p, oid, 16, 0, nullptr, 0, 0, MM, 16, 288, 0);
  gemm(dec16, 288, WchT, 288, bch_p, och, 16, 0, nullptr, 0, 0, MM, 16, 288, 0);
  k_fillxr<<<(MM*8+255)/256, 256, 0, stream>>>(oid, xr16);

  // ---- reg branch ----
  gemm(xr16, 320, regencT, 320, reg_enc_b, nullptr, 0, 0, x16, HH, 0, MM, 256, 320, 1);
  gemm(x16, HH, regWtT, 256, reg_bt, gate, HH, 0, nullptr, 0, 0, MM, 256, 256, 2);
  gemm(x16, HH, regthT, 256, nullptr, t1, HH, 0, nullptr, 0, 0, MM, 256, 256, 0);
  gemm(x16, HH, regWhT, 256, nullptr, fhet, HH, 0, nullptr, 0, 0, MM, 256, 256, 0);
  k_spmm<<<MM, HH, 0, stream>>>(t1, perm, cols, vals, normv, fhom);
  k_combine<<<(MM*HH+255)/256, 256, 0, stream>>>(gate, fhom, fhet, y16, MM*HH);
  gemm(y16, HH, regdecT, 256, reg_dec_b, nullptr, 0, 0, dec16, 288, 26, MM, 256, 256, 1);
  gemm(dec16, 288, WmomT, 288, bmom_p, momb, 16, 0, nullptr, 0, 0, MM, 16, 288, 0);

  // ---- assemble masked output [out_id(8) | charge(1) | mom(3)] ----
  k_final<<<(MM*12+255)/256, 256, 0, stream>>>(oid, och, momb, msk, (float*)d_out);
}
